// AttBlock_51513837748472
// MI455X (gfx1250) — matrix operand order repaired and checked by reading, not compiled, not run on hardware
//
#include <hip/hip_runtime.h>
#include <hip/hip_bf16.h>

// Problem constants from the reference
#define NROW   8192
#define QDIM   512
#define KVDIM  512
#define DK     128

typedef __attribute__((ext_vector_type(16))) __bf16 v16bf;
typedef __attribute__((ext_vector_type(8)))  __bf16 v8bf;
typedef __attribute__((ext_vector_type(4)))  __bf16 v4bf;
typedef __attribute__((ext_vector_type(8)))  float  v8f;

// round-to-nearest-even f32 -> bf16 (integer path; only used in the
// bandwidth-bound conversion passes, never in the GEMM hot loop)
static __device__ __forceinline__ __bf16 f2bf(float f) {
    unsigned u = __builtin_bit_cast(unsigned, f);
    u += 0x7FFFu + ((u >> 16) & 1u);
    unsigned short s = (unsigned short)(u >> 16);
    return __builtin_bit_cast(__bf16, s);
}

// ---------------------------------------------------------------------------
// One-time f32 -> bf16 conversion (memory-bound; n must be a multiple of 4).
// grid*block*4 == n exactly.
// ---------------------------------------------------------------------------
__global__ __launch_bounds__(256) void cvt_f32_to_bf16(
    const float* __restrict__ in, __bf16* __restrict__ out)
{
    const size_t idx = ((size_t)blockIdx.x * blockDim.x + threadIdx.x) * 4;
    float4 f = *(const float4*)(in + idx);
    v4bf o;
    o[0] = f2bf(f.x); o[1] = f2bf(f.y); o[2] = f2bf(f.z); o[3] = f2bf(f.w);
    *(v4bf*)(out + idx) = o;
}

// ---------------------------------------------------------------------------
// GEMM with bias: Y[M, NCOLS] = Xbf[M, K] @ Wbf[NCOLS, K]^T + bias  (f32 out)
// A and B are pre-converted bf16 -> inner loop is pure b128 loads + WMMA.
// One wave -> one 16x16 tile; 8 waves/WG -> 16 rows x 128 cols.
// grid = (M/16, NCOLS/128)
// ---------------------------------------------------------------------------
template <int K, int NCOLS>
__global__ __launch_bounds__(256) void gemm_bias_wmma(
    const __bf16* __restrict__ X, const __bf16* __restrict__ W,
    const float* __restrict__ bias, float* __restrict__ Y)
{
    const int lane = threadIdx.x & 31;
    const int wave = threadIdx.x >> 5;
    const int half = lane >> 4;      // 0: lanes 0-15, 1: lanes 16-31
    const int l16  = lane & 15;

    const int row0 = blockIdx.x * 16;
    const int col0 = blockIdx.y * 128 + wave * 16;

    const int arow = row0 + l16;     // A-matrix row held by this lane
    const int bcol = col0 + l16;     // output column held by this lane

    const __bf16* __restrict__ Xp = X + (size_t)arow * K;
    const __bf16* __restrict__ Wp = W + (size_t)bcol * K;

    v8f acc = {};

    #pragma unroll 4
    for (int k0 = 0; k0 < K; k0 += 32) {
        if (k0 + 32 < K)
            __builtin_prefetch(Xp + k0 + 32, 0, 3);

        // A: 16-bit A 16x32 layout. Lane holds row `arow`:
        //   elems 0-7  at K = k0 + half*8       (8 contiguous bf16 = 16B)
        //   elems 8-15 at K = k0 + 16 + half*8  (8 contiguous bf16 = 16B)
        v8bf a_lo = *(const v8bf*)(Xp + k0 + half * 8);
        v8bf a_hi = *(const v8bf*)(Xp + k0 + 16 + half * 8);
        v16bf a = __builtin_shufflevector(a_lo, a_hi,
            0, 1, 2, 3, 4, 5, 6, 7, 8, 9, 10, 11, 12, 13, 14, 15);

        // B: 16-bit B 32x16 layout. Lane holds column `bcol`:
        //   elems 0-7  at K = k0 + half*8       (8 contiguous bf16 = 16B)
        //   elems 8-15 at K = k0 + 16 + half*8  (8 contiguous bf16 = 16B)
        v8bf b_lo = *(const v8bf*)(Wp + k0 + half * 8);
        v8bf b_hi = *(const v8bf*)(Wp + k0 + 16 + half * 8);
        v16bf b = __builtin_shufflevector(b_lo, b_hi,
            0, 1, 2, 3, 4, 5, 6, 7, 8, 9, 10, 11, 12, 13, 14, 15);

        acc = __builtin_amdgcn_wmma_f32_16x16x32_bf16(
            /*neg_a=*/false, a, /*neg_b=*/false, b,
            /*c_mod=*/(short)0, acc, /*reuse_a=*/false, /*reuse_b=*/false);
    }

    // D layout: VGPR r -> (M = r + half*8, N = l16). NCOLS is compile-time,
    // so the 8 stores become constant IOFFSETs off one base address.
    const float bv = bias[bcol];
    float* __restrict__ Yp = Y + (size_t)(row0 + half * 8) * NCOLS + bcol;
    #pragma unroll
    for (int r = 0; r < 8; ++r)
        Yp[(size_t)r * NCOLS] = acc[r] + bv;
}

// ---------------------------------------------------------------------------
// Fused rank-1 softmax attention middle:
//   x[n,i] = sum_j softmax_j(q[n,i]*k[n,j]/DK) * v[n,j]
// Row max is analytic: m_i = q_i/DK * (q_i>=0 ? max_j k_j : min_j k_j).
// One workgroup (128 threads) per sample; k,v staged in LDS.
// Output written in bf16 (consumed by the proj WMMA GEMM).
// ---------------------------------------------------------------------------
__global__ __launch_bounds__(128) void attn_mid(
    const float* __restrict__ q, const float* __restrict__ k,
    const float* __restrict__ v, __bf16* __restrict__ x)
{
    __shared__ float ks[DK], vs[DK], smax[DK], smin[DK];

    const int n = blockIdx.x;
    const int i = threadIdx.x;

    const float kj = k[(size_t)n * DK + i];
    ks[i]   = kj;
    vs[i]   = v[(size_t)n * DK + i];
    smax[i] = kj;
    smin[i] = kj;
    __syncthreads();

    #pragma unroll
    for (int s = DK / 2; s > 0; s >>= 1) {
        if (i < s) {
            smax[i] = fmaxf(smax[i], smax[i + s]);
            smin[i] = fminf(smin[i], smin[i + s]);
        }
        __syncthreads();
    }
    const float kmax = smax[0];
    const float kmin = smin[0];

    const float qi    = q[(size_t)n * DK + i];
    const float scale = qi * (1.0f / (float)DK);
    const float m     = scale * (qi >= 0.0f ? kmax : kmin);

    float num = 0.0f, den = 0.0f;
    #pragma unroll 8
    for (int j = 0; j < DK; ++j) {
        const float e = __expf(fmaf(scale, ks[j], -m));
        num = fmaf(e, vs[j], num);
        den += e;
    }
    x[(size_t)n * DK + i] = f2bf(num / den);
}

extern "C" void kernel_launch(void* const* d_in, const int* in_sizes, int n_in,
                              void* d_out, int out_size, void* d_ws, size_t ws_size,
                              hipStream_t stream)
{
    const float* x_q  = (const float*)d_in[0];
    const float* x_kv = (const float*)d_in[1];
    const float* Wq_w = (const float*)d_in[2];
    const float* Wq_b = (const float*)d_in[3];
    const float* Wk_w = (const float*)d_in[4];
    const float* Wk_b = (const float*)d_in[5];
    const float* Wv_w = (const float*)d_in[6];
    const float* Wv_b = (const float*)d_in[7];
    const float* Pw   = (const float*)d_in[8];
    const float* Pb   = (const float*)d_in[9];
    float* out = (float*)d_out;

    // ---- workspace layout -------------------------------------------------
    char* w = (char*)d_ws;
    __bf16* xq_bf  = (__bf16*)w;  w += (size_t)NROW * QDIM  * 2;  // 8 MB
    __bf16* xkv_bf = (__bf16*)w;  w += (size_t)NROW * KVDIM * 2;  // 8 MB
    __bf16* wq_bf  = (__bf16*)w;  w += (size_t)DK * QDIM  * 2;
    __bf16* wk_bf  = (__bf16*)w;  w += (size_t)DK * KVDIM * 2;
    __bf16* wv_bf  = (__bf16*)w;  w += (size_t)DK * KVDIM * 2;
    __bf16* pw_bf  = (__bf16*)w;  w += (size_t)KVDIM * DK * 2;
    __bf16* xb_bf  = (__bf16*)w;  w += (size_t)NROW * DK * 2;     // 2 MB
    float*  qf     = (float*)w;   w += (size_t)NROW * DK * 4;     // 4 MB
    float*  kf     = (float*)w;   w += (size_t)NROW * DK * 4;
    float*  vf     = (float*)w;   w += (size_t)NROW * DK * 4;

    const dim3 blk(256);

    // ---- one-time f32 -> bf16 conversions (memory-bound) ------------------
    cvt_f32_to_bf16<<<dim3((NROW * QDIM)  / 1024), blk, 0, stream>>>(x_q,  xq_bf);
    cvt_f32_to_bf16<<<dim3((NROW * KVDIM) / 1024), blk, 0, stream>>>(x_kv, xkv_bf);
    cvt_f32_to_bf16<<<dim3((DK * QDIM)  / 1024), blk, 0, stream>>>(Wq_w, wq_bf);
    cvt_f32_to_bf16<<<dim3((DK * KVDIM) / 1024), blk, 0, stream>>>(Wk_w, wk_bf);
    cvt_f32_to_bf16<<<dim3((DK * KVDIM) / 1024), blk, 0, stream>>>(Wv_w, wv_bf);
    cvt_f32_to_bf16<<<dim3((KVDIM * DK) / 1024), blk, 0, stream>>>(Pw,   pw_bf);

    // ---- q/k/v projections: M=8192, K=512, NCOLS=128 ----------------------
    gemm_bias_wmma<QDIM, DK><<<dim3(NROW / 16, 1), blk, 0, stream>>>(
        xq_bf, wq_bf, Wq_b, qf);
    gemm_bias_wmma<KVDIM, DK><<<dim3(NROW / 16, 1), blk, 0, stream>>>(
        xkv_bf, wk_bf, Wk_b, kf);
    gemm_bias_wmma<KVDIM, DK><<<dim3(NROW / 16, 1), blk, 0, stream>>>(
        xkv_bf, wv_bf, Wv_b, vf);

    // ---- fused rank-1 softmax attention (bf16 output) ----------------------
    attn_mid<<<dim3(NROW), dim3(DK), 0, stream>>>(qf, kf, vf, xb_bf);

    // ---- output projection: M=8192, K=128, NCOLS=512 -----------------------
    gemm_bias_wmma<DK, KVDIM><<<dim3(NROW / 16, KVDIM / 128), blk, 0, stream>>>(
        xb_bf, pw_bf, Pb, out);
}
